// CrossTriplet_30666066494149
// MI455X (gfx1250) — matrix operand order repaired and checked by reading, not compiled, not run on hardware
//
#include <hip/hip_runtime.h>
#include <hip/hip_bf16.h>
#include <math.h>

// ---- CDNA5 WMMA types ----
typedef __attribute__((ext_vector_type(16))) __bf16 v16bf;
typedef __attribute__((ext_vector_type(8)))  float  v8f;

#define FEAT   128
#define MARGIN 0.3f

union Frag { unsigned int u[8]; v16bf v; };

__device__ __forceinline__ unsigned int f2bf(float x) {
    // round-to-nearest-even fp32 -> bf16 (top 16 bits)
    unsigned int u = __builtin_bit_cast(unsigned int, x);
    return (u + 0x7FFFu + ((u >> 16) & 1u)) >> 16;
}
__device__ __forceinline__ float bf2f(unsigned int b) {
    return __builtin_bit_cast(float, b << 16);
}
__device__ __forceinline__ unsigned int pack_hi(float a, float b) {
    return f2bf(a) | (f2bf(b) << 16);
}
__device__ __forceinline__ unsigned int pack_lo(float a, float b) {
    float la = a - bf2f(f2bf(a));
    float lb = b - bf2f(f2bf(b));
    return f2bf(la) | (f2bf(lb) << 16);
}

// ---------------- Kernel 1: row norms + one-time bf16 hi/lo split ----------------
// 8 waves/block, one row per wave; lane handles 4 consecutive features (pairs 2L,2L+1).
__global__ void __launch_bounds__(256) prep_kernel(const float* __restrict__ X,
                                                   float* __restrict__ sq,
                                                   unsigned int* __restrict__ Xhi,
                                                   unsigned int* __restrict__ Xlo, int n) {
    const int wave = threadIdx.x >> 5;
    const int lane = threadIdx.x & 31;
    const int row  = blockIdx.x * 8 + wave;
    if (row >= n) return;
    const float4 v = ((const float4*)(X + (size_t)row * FEAT))[lane]; // 32 lanes * 4 = 128

    // packed bf16 pairs, fragment-friendly row-major [row][64] dword layout
    ((uint2*)(Xhi + (size_t)row * 64))[lane] =
        make_uint2(pack_hi(v.x, v.y), pack_hi(v.z, v.w));
    ((uint2*)(Xlo + (size_t)row * 64))[lane] =
        make_uint2(pack_lo(v.x, v.y), pack_lo(v.z, v.w));

    float s = v.x * v.x + v.y * v.y + v.z * v.z + v.w * v.w;
    #pragma unroll
    for (int m = 16; m; m >>= 1) s += __shfl_xor(s, m, 32);
    if (lane == 0) sq[row] = s;
}

// ---------------- Kernel 2: fused cross-modal hard mining ----------------
// Block = 256 threads = 8 waves; each wave owns one 16-row tile (block covers 128 rows).
// Only the opposite-modality half of columns is visited.
// Mining runs on squared distance (sqrt is monotonic); sqrt applied once per row.
// Double-buffered LDS; staging is a pure b128 copy of precomputed bf16 pairs.
__global__ void __launch_bounds__(256) triplet_tile_kernel(
    const unsigned int* __restrict__ Xhi, const unsigned int* __restrict__ Xlo,
    const int* __restrict__ T, const float* __restrict__ sq,
    float* __restrict__ ap_out, float* __restrict__ an_out, int n) {

    constexpr int LDS_STRIDE = 68;  // 64 dword pairs per row + 4 pad -> conflict-free b128
    __shared__ __align__(16) unsigned int lds_hi[2][16 * LDS_STRIDE];
    __shared__ __align__(16) unsigned int lds_lo[2][16 * LDS_STRIDE];

    const int tid  = threadIdx.x;
    const int lane = tid & 31;
    const int wave = tid >> 5;
    const int h    = lane >> 4;    // half-wave select
    const int l15  = lane & 15;
    const int half = n >> 1;

    const int rowBase = (blockIdx.x * 8 + wave) * 16;
    const int aRow    = rowBase + l15;   // A-matrix row held by this lane

    // ---- persistent A fragments: direct dword gathers from precomputed hi/lo ----
    Frag Ahi[4], Alo[4];
    {
        const unsigned int* hr = Xhi + (size_t)aRow * 64;
        const unsigned int* lr = Xlo + (size_t)aRow * 64;
        #pragma unroll
        for (int c = 0; c < 4; ++c) {
            #pragma unroll
            for (int v = 0; v < 8; ++v) {
                // pair index of K = 32c + 16h + 2v
                const int p = 16 * c + 8 * h + v;
                Ahi[c].u[v] = hr[p];
                Alo[c].u[v] = lr[p];
            }
        }
    }

    // ---- per-row metadata (C layout: VGPR r -> row r + 8*h, col l15) ----
    float sq_row[8]; int t_row[8];
    #pragma unroll
    for (int r = 0; r < 8; ++r) {
        const int row = rowBase + r + 8 * h;
        sq_row[r] = sq[row];
        t_row[r]  = T[row];
    }

    float ap[8], an[8];   // running max/min of SQUARED distance
    #pragma unroll
    for (int r = 0; r < 8; ++r) { ap[r] = -INFINITY; an[r] = INFINITY; }

    // visit only the opposite-modality half of the columns
    const int colStart = (rowBase < half) ? half : 0;
    const int nTiles   = half >> 4;

    const int sRow = tid >> 4;   // staging: row within col tile (0..15)
    const int sJ   = tid & 15;   // staging: 4-dword chunk within the row
    const unsigned int sBase = sRow * LDS_STRIDE + 4 * sJ;
    const size_t sOff = (size_t)sRow * 64 + 4 * sJ;

    // ---- prologue: stage tile 0 into buffer 0 (pure copy) ----
    {
        *(uint4*)&lds_hi[0][sBase] = *(const uint4*)(Xhi + (size_t)colStart * 64 + sOff);
        *(uint4*)&lds_lo[0][sBase] = *(const uint4*)(Xlo + (size_t)colStart * 64 + sOff);
    }
    __syncthreads();

    for (int jt = 0; jt < nTiles; ++jt) {
        const int buf     = jt & 1;
        const int colBase = colStart + (jt << 4);

        // issue next tile's global loads early (latency hidden under WMMAs)
        const bool haveNext = (jt + 1 < nTiles);
        uint4 nh, nl;
        if (haveNext) {
            const size_t nbase = (size_t)(colBase + 16) * 64 + sOff;
            nh = *(const uint4*)(Xhi + nbase);
            nl = *(const uint4*)(Xlo + nbase);
        }

        // ---- 16x16 tile of G = X.X^T via bf16x3 split (12 WMMAs) ----
        v8f acc = {};
        #pragma unroll
        for (int c = 0; c < 4; ++c) {
            Frag Bhi, Blo;
            const unsigned int base = l15 * LDS_STRIDE + 16 * c + 8 * h;
            const uint4 h0 = *(const uint4*)&lds_hi[buf][base];
            const uint4 h1 = *(const uint4*)&lds_hi[buf][base + 4];
            const uint4 l0 = *(const uint4*)&lds_lo[buf][base];
            const uint4 l1 = *(const uint4*)&lds_lo[buf][base + 4];
            Bhi.u[0] = h0.x; Bhi.u[1] = h0.y; Bhi.u[2] = h0.z; Bhi.u[3] = h0.w;
            Bhi.u[4] = h1.x; Bhi.u[5] = h1.y; Bhi.u[6] = h1.z; Bhi.u[7] = h1.w;
            Blo.u[0] = l0.x; Blo.u[1] = l0.y; Blo.u[2] = l0.z; Blo.u[3] = l0.w;
            Blo.u[4] = l1.x; Blo.u[5] = l1.y; Blo.u[6] = l1.z; Blo.u[7] = l1.w;

            acc = __builtin_amdgcn_wmma_f32_16x16x32_bf16(false, Ahi[c].v, false, Bhi.v,
                                                          (short)0, acc, false, false);
            acc = __builtin_amdgcn_wmma_f32_16x16x32_bf16(false, Ahi[c].v, false, Blo.v,
                                                          (short)0, acc, false, false);
            acc = __builtin_amdgcn_wmma_f32_16x16x32_bf16(false, Alo[c].v, false, Bhi.v,
                                                          (short)0, acc, false, false);
        }

        // ---- epilogue: mine on squared distances (all pairs here are cross-modal) ----
        const int   col = colBase + l15;
        const float sqc = sq[col];
        const int   tc  = T[col];
        #pragma unroll
        for (int r = 0; r < 8; ++r) {
            const float d2 = __builtin_fmaf(-2.0f, acc[r], sq_row[r] + sqc);
            if (t_row[r] == tc) ap[r] = fmaxf(ap[r], d2);   // positive: same id
            else                an[r] = fminf(an[r], d2);   // negative: diff id
        }

        // ---- store next tile into the other buffer (pure copy) ----
        if (haveNext) {
            const int nb = buf ^ 1;
            *(uint4*)&lds_hi[nb][sBase] = nh;
            *(uint4*)&lds_lo[nb][sBase] = nl;
        }
        __syncthreads();
    }

    // ---- reduce across the 16 lanes holding the same row ----
    #pragma unroll
    for (int r = 0; r < 8; ++r) {
        #pragma unroll
        for (int m = 1; m < 16; m <<= 1) {
            ap[r] = fmaxf(ap[r], __shfl_xor(ap[r], m, 16));
            an[r] = fminf(an[r], __shfl_xor(an[r], m, 16));
        }
    }
    if (l15 == 0) {
        #pragma unroll
        for (int r = 0; r < 8; ++r) {
            const int row = rowBase + r + 8 * h;
            ap_out[row] = sqrtf(fmaxf(ap[r], 1e-12f));   // sqrt once per row
            an_out[row] = sqrtf(fmaxf(an[r], 1e-12f));
        }
    }
}

// ---------------- Kernel 3: margin-ranking loss reduction ----------------
__global__ void __launch_bounds__(256) loss_kernel(const float* __restrict__ ap,
                                                   const float* __restrict__ an,
                                                   float* __restrict__ out, int n) {
    __shared__ float ssum[256];
    __shared__ int   scnt[256];
    const int tid = threadIdx.x;
    float s = 0.0f; int c = 0;
    for (int i = tid; i < n; i += 256) {
        const float a = ap[i], b = an[i];
        const float v = a - b + MARGIN;
        s += (v > 0.0f) ? v : 0.0f;
        c += (b >= a) ? 1 : 0;
    }
    ssum[tid] = s; scnt[tid] = c;
    __syncthreads();
    for (int m = 128; m; m >>= 1) {
        if (tid < m) { ssum[tid] += ssum[tid + m]; scnt[tid] += scnt[tid + m]; }
        __syncthreads();
    }
    if (tid == 0) {
        out[0] = ssum[0] / (float)n;   // loss
        out[1] = (float)scnt[0];       // correct count
    }
}

extern "C" void kernel_launch(void* const* d_in, const int* in_sizes, int n_in,
                              void* d_out, int out_size, void* d_ws, size_t ws_size,
                              hipStream_t stream) {
    const float* X = (const float*)d_in[0];   // [n,128] fp32
    const int*   T = (const int*)d_in[1];     // [n] int32
    const int    n = in_sizes[1];             // 16384

    float* ws = (float*)d_ws;
    float* sq = ws;            // n floats
    float* ap = ws + n;        // n floats
    float* an = ws + 2 * n;    // n floats
    unsigned int* Xhi = (unsigned int*)(ws + 3 * n);       // n*64 dwords (bf16 hi pairs)
    unsigned int* Xlo = Xhi + (size_t)n * 64;              // n*64 dwords (bf16 lo pairs)

    prep_kernel<<<n / 8, 256, 0, stream>>>(X, sq, Xhi, Xlo, n);
    triplet_tile_kernel<<<n / 128, 256, 0, stream>>>(Xhi, Xlo, T, sq, ap, an, n);
    loss_kernel<<<1, 256, 0, stream>>>(ap, an, (float*)d_out, n);
}
